// AcnerSeq2Seq_56607668961825
// MI455X (gfx1250) — matrix operand order repaired and checked by reading, not compiled, not run on hardware
//
#include <hip/hip_runtime.h>
#include <hip/hip_bf16.h>

// ---------------------------------------------------------------------------
// Seq2Seq LSTM (encoder+decoder, shared cell) for MI455X / gfx1250.
//   * Weights packed once to bf16 WMMA-B-fragment tiles.
//   * ONE persistent kernel runs all 512 sequential timesteps:
//       - per-WG weight slab (192 KB) staged into LDS ONCE via TDM
//         (tensor_load_to_lds + s_wait_tensorcnt), reused for all steps
//       - cell state c held in VGPRs for the whole recurrence
//       - steps separated by device-scope atomic arrive/spin barrier
//   * WMMA projection + wave-level softmax.
// ---------------------------------------------------------------------------

typedef __bf16 bf16;
typedef __attribute__((ext_vector_type(16))) __bf16 v16bf;
typedef __attribute__((ext_vector_type(8)))  __bf16 v8bf;
typedef __attribute__((ext_vector_type(8)))  float  v8f;
typedef __attribute__((ext_vector_type(4))) unsigned int u32x4;
typedef __attribute__((ext_vector_type(8))) int i32x8;
typedef __attribute__((ext_vector_type(4))) int i32x4;

union AFrag { v16bf v; v8bf h[2]; };

#define B_SZ 64
#define T_SZ 256
#define E_SZ 512
#define H_SZ 1024
#define VT   64
#define KTOT (E_SZ + H_SZ)   /* 1536 */
#define N4H  (4 * H_SZ)      /* 4096 */
#define KT_ALL 48            /* KTOT/32 */
#define KT_X   16            /* E/32   */
#define NBLK 64              /* persistent workgroups */

#if __has_builtin(__builtin_amdgcn_tensor_load_to_lds)
#define HAVE_TDM 1
#else
#define HAVE_TDM 0
#endif

__device__ __forceinline__ float sigmoidf_(float x) {
  return 1.0f / (1.0f + __expf(-x));
}
__device__ __forceinline__ float tanhf_(float x) {
  x = fminf(fmaxf(x, -15.0f), 15.0f);
  float e = __expf(2.0f * x);
  return (e - 1.0f) / (e + 1.0f);
}

// ---------------------------------------------------------------------------
// Embedding gather + f32->bf16.  out layout [T, B, E]; tok layout [B, T].
// ---------------------------------------------------------------------------
__global__ __launch_bounds__(256) void k_embed(const int* __restrict__ tok,
                                               const float* __restrict__ emb,
                                               bf16* __restrict__ out)
{
  size_t gid = (size_t)blockIdx.x * blockDim.x + threadIdx.x;   // T*B*E total
  int e      = (int)(gid & (E_SZ - 1));
  size_t tb  = gid >> 9;
  int b      = (int)(tb & (B_SZ - 1));
  int t      = (int)(tb >> 6);
  int token  = tok[b * T_SZ + t];
  out[gid] = (bf16)emb[(size_t)token * E_SZ + e];
}

// ---------------------------------------------------------------------------
// Pack row-major f32 W[K,N] into bf16 WMMA B-fragment tiles (32x16 per tile,
// 512 bf16 contiguous; lane*16 gives each lane a contiguous 32B fragment).
// ---------------------------------------------------------------------------
__global__ __launch_bounds__(256) void k_pack_w(const float* __restrict__ W,
                                                bf16* __restrict__ out,
                                                int K, int N)
{
  size_t gid = (size_t)blockIdx.x * blockDim.x + threadIdx.x;
  if (gid >= (size_t)K * N) return;
  int e       = (int)(gid & 15);
  int lane    = (int)((gid >> 4) & 31);
  size_t tile = gid >> 9;
  int KT = K >> 5;
  int kt = (int)(tile % KT);
  int nt = (int)(tile / KT);
  int k = kt * 32 + ((e & 8) << 1) + ((lane >> 4) << 3) + (e & 7);
  int n = nt * 16 + (lane & 15);
  out[gid] = (bf16)W[(size_t)k * N + n];
}

// ---------------------------------------------------------------------------
// Device-wide barrier: per-block arrive on an agent-scope atomic counter,
// block 0..63; monotonic target so it is reusable across all 512 steps.
// ---------------------------------------------------------------------------
__device__ __forceinline__ void grid_barrier(unsigned* cnt, int step)
{
  __syncthreads();
  if (threadIdx.x == 0) {
    __hip_atomic_fetch_add(cnt, 1u, __ATOMIC_RELEASE, __HIP_MEMORY_SCOPE_AGENT);
    unsigned target = (unsigned)(step + 1) * NBLK;
    while (__hip_atomic_load(cnt, __ATOMIC_ACQUIRE, __HIP_MEMORY_SCOPE_AGENT) < target) {
      __builtin_amdgcn_s_sleep(2);
    }
  }
  __syncthreads();
  __threadfence();   // acquire: invalidate local caches before reading new h
}

// ---------------------------------------------------------------------------
// Persistent LSTM: all 512 timesteps (256 encoder + 256 decoder).
// Grid: 64 blocks x 128 threads (4 waves).  Block -> 16 hidden columns,
// wave -> 16 batch rows.  Weight slab (4 gates x 48 ktiles x 1KB = 192KB)
// staged into LDS once via TDM, reused every step.  c stays in VGPRs.
// ---------------------------------------------------------------------------
__global__ __launch_bounds__(128) void k_lstm_all(
    const bf16*  __restrict__ xs,     // [256, 64, 512]  encoder inputs
    const bf16*  __restrict__ ys,     // [256, 64, 512]  decoder inputs
    const bf16*  __restrict__ Wpk,    // packed tiles (K=1536, N=4096)
    const float* __restrict__ bias,   // [4096]
    bf16*        __restrict__ hb0,    // [64,1024] ping (zero-initialized)
    bf16*        __restrict__ hb1,    // [64,1024] pong
    bf16*        __restrict__ hsAll,  // [256, 64, 1024] decoder outputs
    unsigned*    __restrict__ barCnt) // zero-initialized
{
  extern __shared__ char smem[];
  bf16* Bs = (bf16*)smem;             // gate g, ktile kt at (g*48+kt)*512 elems

  const int lane     = threadIdx.x & 31;
  const int wave     = threadIdx.x >> 5;
  const int h0       = blockIdx.x << 4;
  const int m0       = wave << 4;
  const int laneHalf = lane >> 4;
  const int row      = m0 + (lane & 15);

  // ---- Stage this block's weight slab into LDS (once) -------------------
  // Gate g slab: tiles [(g*64+blk)*48, +48) = 48KB contiguous; gate stride
  // 64*48KB = 3MB.  2D TDM copy: 4 rows of 6144 8-byte elements.
#if HAVE_TDM
  if (wave == 0) {   // one wave issues the DMA (TDM ignores EXEC)
    unsigned lds_base = (unsigned)(size_t)smem;            // LDS byte offset
    unsigned long long ga =
        (unsigned long long)(size_t)(Wpk + (size_t)blockIdx.x * KT_ALL * 512);
    u32x4 g0;
    g0[0] = 1u;                                            // count=1 (valid D#)
    g0[1] = lds_base;                                      // lds_addr
    g0[2] = (unsigned)(ga & 0xFFFFFFFFull);                // global_addr[31:0]
    g0[3] = (unsigned)((ga >> 32) & 0x1FFFFFFull) | (2u << 30); // [56:32]|type=2
    i32x8 g1;
    g1[0] = (3 << 16);                  // workgroup_mask=0, data_size=3 (8B)
    g1[1] = (int)((6144u & 0xFFFFu) << 16);   // tensor_dim0[15:0]
    g1[2] = (int)((6144u >> 16) | (4u << 16)); // dim0[31:16] | tensor_dim1[15:0]=4
    g1[3] = (int)(6144u << 16);         // tensor_dim1[31:16]=0 | tile_dim0=6144
    g1[4] = 4;                          // tile_dim1=4, tile_dim2=0
    g1[5] = 393216;                     // tensor_dim0_stride[31:0] (8B units)
    g1[6] = 0;                          // stride0[47:32] | dim1_stride[15:0]
    g1[7] = 0;
    i32x4 gz4 = {0, 0, 0, 0};
    i32x8 gz8 = {0, 0, 0, 0, 0, 0, 0, 0};
    __builtin_amdgcn_tensor_load_to_lds(g0, g1, gz4, gz4, gz8, 0);
    __builtin_amdgcn_s_wait_tensorcnt(0);
  }
  __syncthreads();
#else
  // Fallback: cooperative 16B-chunk staging.
  {
    const v8bf* wsrc = (const v8bf*)Wpk;
    v8bf* dst = (v8bf*)Bs;
    for (int i = threadIdx.x; i < 4 * KT_ALL * 64; i += 128) {
      int g   = i / (KT_ALL * 64);
      int rem = i - g * (KT_ALL * 64);
      dst[g * (KT_ALL * 64) + rem] =
          wsrc[((size_t)(g * NBLK + blockIdx.x) * KT_ALL) * 64 + rem];
    }
    __syncthreads();
  }
#endif

  // ---- Per-lane invariants ----------------------------------------------
  const int hcol = h0 + (lane & 15);
  const float bi = bias[0 * H_SZ + hcol];
  const float bj = bias[1 * H_SZ + hcol];
  const float bfo = bias[2 * H_SZ + hcol] + 1.0f;   // + forget_bias
  const float bo = bias[3 * H_SZ + hcol];

  float creg[8];
#pragma unroll
  for (int r = 0; r < 8; ++r) creg[r] = 0.0f;

  // ---- 512 sequential timesteps -----------------------------------------
  for (int t = 0; t < 2 * T_SZ; ++t) {
    const bf16* abase = (t < T_SZ)
        ? (xs + (size_t)t * (B_SZ * E_SZ))
        : (ys + (size_t)(t - T_SZ) * (B_SZ * E_SZ));
    const bf16* hprev = (t & 1) ? hb1 : hb0;
    bf16*       hnext = (t & 1) ? hb0 : hb1;

    v8f zero = {};
    v8f acc[4] = { zero, zero, zero, zero };

    for (int kt = 0; kt < KT_ALL; ++kt) {
      AFrag a;
      if (kt < KT_X) {
        const bf16* ar = abase + (size_t)row * E_SZ + (kt << 5) + (laneHalf << 3);
        a.h[0] = *(const v8bf*)(ar);
        a.h[1] = *(const v8bf*)(ar + 16);
      } else {
        const bf16* ar = hprev + (size_t)row * H_SZ + ((kt - KT_X) << 5) + (laneHalf << 3);
        a.h[0] = *(const v8bf*)(ar);
        a.h[1] = *(const v8bf*)(ar + 16);
      }
#pragma unroll
      for (int g = 0; g < 4; ++g) {
        const v16bf bfrag =
            *(const v16bf*)(Bs + (((g * KT_ALL) + kt) << 9) + (lane << 4));
        acc[g] = __builtin_amdgcn_wmma_f32_16x16x32_bf16(
            false, a.v, false, bfrag, (short)0, acc[g], false, false);
      }
    }

    // Gate fusion: C-tile VGPR r holds (M = r + 8*laneHalf, N = lane&15),
    // identical coords across the 4 gate accumulators.
#pragma unroll
    for (int r = 0; r < 8; ++r) {
      int m = m0 + r + (laneHalf << 3);
      size_t idx = (size_t)m * H_SZ + hcol;
      float gi = acc[0][r] + bi;
      float gj = acc[1][r] + bj;
      float gf = acc[2][r] + bfo;
      float go = acc[3][r] + bo;
      creg[r] = sigmoidf_(gf) * creg[r] + sigmoidf_(gi) * tanhf_(gj);
      float hnew = sigmoidf_(go) * tanhf_(creg[r]);
      bf16 hb = (bf16)hnew;
      hnext[idx] = hb;
      if (t >= T_SZ) hsAll[(size_t)(t - T_SZ) * (B_SZ * H_SZ) + idx] = hb;
    }

    grid_barrier(barCnt, t);
  }
}

// ---------------------------------------------------------------------------
// Projection: logits[m, n] = hs[m, :] @ W_proj[:, n] + b_proj[n]
// M = 16384, K = 1024, N = 64.  1024 blocks x 4 waves.
// ---------------------------------------------------------------------------
__global__ __launch_bounds__(128) void k_proj(
    const bf16*  __restrict__ hs,
    const bf16*  __restrict__ Wpk,
    const float* __restrict__ bproj,
    float*       __restrict__ logits)
{
  const int lane     = threadIdx.x & 31;
  const int wave     = threadIdx.x >> 5;
  const int m0       = blockIdx.x << 4;
  const int laneHalf = lane >> 4;
  const int row      = m0 + (lane & 15);
  const int KT       = H_SZ / 32;

  v8f acc = {};
  for (int kt = 0; kt < KT; ++kt) {
    AFrag a;
    const bf16* ar = hs + (size_t)row * H_SZ + (kt << 5) + (laneHalf << 3);
    a.h[0] = *(const v8bf*)(ar);
    a.h[1] = *(const v8bf*)(ar + 16);
    const v16bf bfrag =
        *(const v16bf*)(Wpk + (((size_t)(wave * KT + kt)) << 9) + (lane << 4));
    acc = __builtin_amdgcn_wmma_f32_16x16x32_bf16(
        false, a.v, false, bfrag, (short)0, acc, false, false);
  }
  int n = (wave << 4) + (lane & 15);
  float bpv = bproj[n];
#pragma unroll
  for (int r = 0; r < 8; ++r) {
    int m = m0 + r + (laneHalf << 3);
    logits[(size_t)m * VT + n] = acc[r] + bpv;
  }
}

// ---------------------------------------------------------------------------
// Softmax over V=64, one wave32 per row, 2 cols/lane + shfl_xor reductions.
// ---------------------------------------------------------------------------
__global__ __launch_bounds__(256) void k_softmax(const float* __restrict__ logits,
                                                 float* __restrict__ out)
{
  int lane = threadIdx.x & 31;
  int row  = blockIdx.x * 8 + (threadIdx.x >> 5);   // t*64 + b
  const float* lr = logits + (size_t)row * VT;
  float v0 = lr[lane];
  float v1 = lr[lane + 32];
  float mx = fmaxf(v0, v1);
#pragma unroll
  for (int m = 16; m; m >>= 1) mx = fmaxf(mx, __shfl_xor(mx, m, 32));
  float e0 = __expf(v0 - mx);
  float e1 = __expf(v1 - mx);
  float s = e0 + e1;
#pragma unroll
  for (int m = 16; m; m >>= 1) s += __shfl_xor(s, m, 32);
  float inv = 1.0f / s;
  int t = row >> 6;
  int b = row & (B_SZ - 1);
  float* orow = out + (size_t)b * (T_SZ * VT) + (size_t)t * VT;
  orow[lane]      = e0 * inv;
  orow[lane + 32] = e1 * inv;
}

// ---------------------------------------------------------------------------
extern "C" void kernel_launch(void* const* d_in, const int* in_sizes, int n_in,
                              void* d_out, int out_size, void* d_ws, size_t ws_size,
                              hipStream_t stream)
{
  const int*   src  = (const int*)d_in[0];
  const int*   tgt  = (const int*)d_in[1];
  const float* embS = (const float*)d_in[2];
  const float* embT = (const float*)d_in[3];
  const float* Wl   = (const float*)d_in[4];
  const float* bl   = (const float*)d_in[5];
  const float* Wp   = (const float*)d_in[6];
  const float* bp   = (const float*)d_in[7];
  float* out = (float*)d_out;

  char* ws = (char*)d_ws;
  size_t off = 0;
  auto alloc = [&](size_t bytes) -> char* {
    char* p = ws + off;
    off = (off + bytes + 255) & ~(size_t)255;
    return p;
  };

  bf16*     xs    = (bf16*)alloc((size_t)T_SZ * B_SZ * E_SZ * 2);
  bf16*     ys    = (bf16*)alloc((size_t)T_SZ * B_SZ * E_SZ * 2);
  bf16*     Wlpk  = (bf16*)alloc((size_t)KTOT * N4H * 2);
  bf16*     Wppk  = (bf16*)alloc((size_t)H_SZ * VT * 2);
  bf16*     hA    = (bf16*)alloc((size_t)B_SZ * H_SZ * 2);
  bf16*     hB    = (bf16*)alloc((size_t)B_SZ * H_SZ * 2);
  bf16*     hsAll = (bf16*)alloc((size_t)T_SZ * B_SZ * H_SZ * 2);
  float*    lg    = (float*)alloc((size_t)T_SZ * B_SZ * VT * 4);
  unsigned* bar   = (unsigned*)alloc(256);

  (void)hipMemsetAsync(hA,  0, (size_t)B_SZ * H_SZ * 2, stream);
  (void)hipMemsetAsync(bar, 0, 256, stream);

  k_embed<<<(T_SZ * B_SZ * E_SZ) / 256, 256, 0, stream>>>(src, embS, xs);
  k_embed<<<(T_SZ * B_SZ * E_SZ) / 256, 256, 0, stream>>>(tgt, embT, ys);
  k_pack_w<<<(KTOT * N4H) / 256, 256, 0, stream>>>(Wl, Wlpk, KTOT, N4H);
  k_pack_w<<<(H_SZ * VT) / 256, 256, 0, stream>>>(Wp, Wppk, H_SZ, VT);

  // One persistent kernel for all 512 sequential steps.
  // Dynamic LDS: 4 gates * 48 ktiles * 1KB = 192KB (WGP has 320KB).
  k_lstm_all<<<NBLK, 128, 4 * KT_ALL * 1024, stream>>>(
      xs, ys, Wlpk, bl, hA, hB, hsAll, bar);

  k_proj<<<(T_SZ * B_SZ) / 16, 128, 0, stream>>>(hsAll, Wppk, bp, lg);
  k_softmax<<<(T_SZ * B_SZ) / 8, 256, 0, stream>>>(lg, out);
}
